// FlowEmbedding_49185965474180
// MI455X (gfx1250) — matrix operand order repaired and checked by reading, not compiled, not run on hardware
//
#include <hip/hip_runtime.h>
#include <hip/hip_bf16.h>

// ---------------- problem constants ----------------
#define NPTS   8192
#define CFEAT  128
#define KNN_K  16
#define NK     (NPTS * KNN_K)       // 131072 GEMM rows
#define CIN0   259                  // 3 + 128 + 128
#define KPAD0  288                  // 9 * 32
#define BN_EPS 1e-5f
#define SLOPE  0.01f

typedef __attribute__((ext_vector_type(16))) __bf16        v16bf;
typedef __attribute__((ext_vector_type(8)))  float         v8f;
typedef __attribute__((ext_vector_type(4)))  unsigned int  u32x4;

union ABfrag { u32x4 u[2]; v16bf v; };

// ---------------- workspace layout ----------------
#define OFF_IDX   ((size_t)0)                       // 131072 * 4        = 524288
#define OFF_STATS ((size_t)524288)                  // 4 * 256 * 4       = 4096
#define OFF_W     ((size_t)528384)                  // up to 128*288*2   = 73728
#define OFF_X     ((size_t)602112)                  // 131072*288*2      = 75497472
#define OFF_Y     ((size_t)76099584)                // 131072*256*4      = 134217728

// ============================================================
// KNN: one thread per query point; pos2 tiled through LDS.
// Register top-16 kept sorted with fully-unrolled compare-swaps.
// ============================================================
__global__ __launch_bounds__(256) void knn_kernel(const float* __restrict__ pos1,
                                                  const float* __restrict__ pos2,
                                                  int* __restrict__ idxb) {
    const int q = blockIdx.x * 256 + threadIdx.x;
    const float px = pos1[q * 3 + 0];
    const float py = pos1[q * 3 + 1];
    const float pz = pos1[q * 3 + 2];

    float bd[KNN_K];
    int   bi[KNN_K];
#pragma unroll
    for (int i = 0; i < KNN_K; ++i) { bd[i] = 3.402823466e38f; bi[i] = 0; }

    __shared__ float sx[256], sy[256], sz[256];

    for (int tile = 0; tile < NPTS / 256; ++tile) {
        const int p = tile * 256 + threadIdx.x;
        sx[threadIdx.x] = pos2[p * 3 + 0];
        sy[threadIdx.x] = pos2[p * 3 + 1];
        sz[threadIdx.x] = pos2[p * 3 + 2];
        __syncthreads();
#pragma unroll 4
        for (int j = 0; j < 256; ++j) {
            const float dx = sx[j] - px;
            const float dy = sy[j] - py;
            const float dz = sz[j] - pz;
            const float d2 = dx * dx + dy * dy + dz * dz;
            if (d2 < bd[KNN_K - 1]) {
                bd[KNN_K - 1] = d2;
                bi[KNN_K - 1] = tile * 256 + j;
#pragma unroll
                for (int s = KNN_K - 1; s > 0; --s) {
                    if (bd[s] < bd[s - 1]) {
                        float td = bd[s]; bd[s] = bd[s - 1]; bd[s - 1] = td;
                        int   ti = bi[s]; bi[s] = bi[s - 1]; bi[s - 1] = ti;
                    }
                }
            }
        }
        __syncthreads();
    }
#pragma unroll
    for (int k = 0; k < KNN_K; ++k) idxb[q * KNN_K + k] = bi[k];
}

// ============================================================
// Build layer-0 input: (NK, 288) bf16, rows = (n,k) pairs,
// cols = [pos_diff(3) | feat2[idx](128) | feat1(128) | pad(29)=0]
// ============================================================
__global__ __launch_bounds__(256) void build_x0_kernel(const float* __restrict__ pos1,
                                                       const float* __restrict__ pos2,
                                                       const float* __restrict__ feat1,
                                                       const float* __restrict__ feat2,
                                                       const int* __restrict__ idxb,
                                                       __hip_bfloat16* __restrict__ X) {
    const long long g = (long long)blockIdx.x * 256 + threadIdx.x;
    const int row = (int)(g / KPAD0);
    const int col = (int)(g % KPAD0);
    const int n = row >> 4;
    const int id = idxb[row];
    float v;
    if (col < 3)            v = pos2[id * 3 + col] - pos1[n * 3 + col];
    else if (col < 3 + CFEAT)       v = feat2[id * CFEAT + (col - 3)];
    else if (col < 3 + 2 * CFEAT)   v = feat1[n * CFEAT + (col - 3 - CFEAT)];
    else                    v = 0.0f;
    X[g] = __float2bfloat16(v);
}

// ============================================================
// Convert weights (COUT, CIN) f32 -> (COUT, KPAD) bf16, zero padded
// ============================================================
__global__ __launch_bounds__(256) void convert_w_kernel(const float* __restrict__ W,
                                                        __hip_bfloat16* __restrict__ Wb,
                                                        int cin, int kpad) {
    const int g = blockIdx.x * 256 + threadIdx.x;
    const int c = g / kpad;
    const int k = g % kpad;
    const float v = (k < cin) ? W[c * cin + k] : 0.0f;
    Wb[g] = __float2bfloat16(v);
}

// ============================================================
// bf16 WMMA GEMM: Y(NK, COUT) f32 = X(NK, KPAD) @ Wb(COUT, KPAD)^T
// Block tile 64(M) x 128(N); 8 waves, each wave a 32x32 region
// = 2x2 v_wmma_f32_16x16x32_bf16 tiles. LDS-staged operands,
// fragment reads follow ISA 7.12.2 VGPR layouts.
// ============================================================
template <int KSTEPS, int COUT>
__global__ __launch_bounds__(256) void gemm_bf16_kernel(const u32x4* __restrict__ X,
                                                        const u32x4* __restrict__ Wb,
                                                        float* __restrict__ Y) {
    constexpr int KPAD = KSTEPS * 32;
    __shared__ unsigned short shA[64 * 40];    // 64 rows, pitch 40 bf16 (80B, b128-aligned)
    __shared__ unsigned short shB[128 * 40];   // 128 cols of W, pitch 40 bf16

    const int rowBase = blockIdx.x * 64;
    const int nBase   = blockIdx.y * 128;
    const int t    = threadIdx.x;
    const int wave = t >> 5;
    const int lane = t & 31;
    const int wm   = wave & 1;        // 2 M-subtiles of 32
    const int wn   = wave >> 1;       // 4 N-subtiles of 32
    const int r15  = lane & 15;
    const int h    = lane >> 4;

    const int la_r = t >> 2;                  // A loader: row 0..63
    const int la_k = (t & 3) * 8;             // 8 bf16 (one b128) each
    const int lb_c = t >> 2;                  // B loader: W row 0..63 (+64 second pass)
    const int lb_k = (t & 3) * 8;

    v8f acc[2][2] = {};

    for (int kt = 0; kt < KSTEPS; ++kt) {
        __syncthreads();
        // stage A tile (64x32 bf16)
        {
            u32x4 av = X[((size_t)(rowBase + la_r) * KPAD + kt * 32 + la_k) >> 3];
            *(u32x4*)&shA[la_r * 40 + la_k] = av;
        }
        // stage B tile (128x32 bf16), two passes of 64 rows
#pragma unroll
        for (int i = 0; i < 2; ++i) {
            const int c = lb_c + i * 64;
            u32x4 bv = Wb[((size_t)(nBase + c) * KPAD + kt * 32 + lb_k) >> 3];
            *(u32x4*)&shB[c * 40 + lb_k] = bv;
        }
        __syncthreads();

        if (kt + 1 < KSTEPS)
            __builtin_prefetch(&X[((size_t)(rowBase + la_r) * KPAD + (kt + 1) * 32 + la_k) >> 3], 0, 3);

        // A fragments: lane(row)=r15, half h selects K 0-7/16-23 vs 8-15/24-31
        ABfrag a[2], b[2];
#pragma unroll
        for (int mi = 0; mi < 2; ++mi) {
            const int base = (wm * 32 + mi * 16 + r15) * 40 + h * 8;
            a[mi].u[0] = *(const u32x4*)&shA[base];
            a[mi].u[1] = *(const u32x4*)&shA[base + 16];
        }
        // B fragments: lane(col)=r15, half h selects K 0-7/16-23 vs 8-15/24-31
#pragma unroll
        for (int ni = 0; ni < 2; ++ni) {
            const int base = (wn * 32 + ni * 16 + r15) * 40 + h * 8;
            b[ni].u[0] = *(const u32x4*)&shB[base];
            b[ni].u[1] = *(const u32x4*)&shB[base + 16];
        }
#pragma unroll
        for (int mi = 0; mi < 2; ++mi)
#pragma unroll
            for (int ni = 0; ni < 2; ++ni)
                acc[mi][ni] = __builtin_amdgcn_wmma_f32_16x16x32_bf16(
                    false, a[mi].v, false, b[ni].v, (short)0, acc[mi][ni], false, false);
    }

    // C layout: VGPR g -> M=g (lanes 0-15), M=g+8 (lanes 16-31); N=lane&15
#pragma unroll
    for (int mi = 0; mi < 2; ++mi) {
#pragma unroll
        for (int ni = 0; ni < 2; ++ni) {
            const int col = nBase + wn * 32 + ni * 16 + r15;
            const int m0  = rowBase + wm * 32 + mi * 16 + h * 8;
#pragma unroll
            for (int vg = 0; vg < 8; ++vg)
                Y[(size_t)(m0 + vg) * COUT + col] = acc[mi][ni][vg];
        }
    }
}

// ============================================================
// BN statistics: per-channel sum / sumsq over NK rows
// ============================================================
__global__ __launch_bounds__(256) void zero_stats_kernel(float* gsum, float* gsumsq) {
    gsum[threadIdx.x] = 0.0f;
    gsumsq[threadIdx.x] = 0.0f;
}

__global__ __launch_bounds__(256) void stats_kernel(const float* __restrict__ Y,
                                                    float* __restrict__ gsum,
                                                    float* __restrict__ gsumsq,
                                                    int cout) {
    const int t = threadIdx.x;
    const int rpi = 256 / cout;               // rows handled per iteration
    const int c = t & (cout - 1);
    const int rsub = t / cout;
    const int rowsPerBlock = NK / 256;        // 512
    const int r0 = blockIdx.x * rowsPerBlock;
    float s = 0.0f, s2 = 0.0f;
    for (int r = rsub; r < rowsPerBlock; r += rpi) {
        const float y = Y[(size_t)(r0 + r) * cout + c];
        s += y;
        s2 += y * y;
    }
    __shared__ float ls[256], ls2[256];
    ls[t] = s; ls2[t] = s2;
    __syncthreads();
    if (t < cout) {
        float a = ls[t], a2 = ls2[t];
        for (int o = t + cout; o < 256; o += cout) { a += ls[o]; a2 += ls2[o]; }
        atomicAdd(&gsum[t], a);
        atomicAdd(&gsumsq[t], a2);
    }
}

__global__ __launch_bounds__(256) void finalize_stats_kernel(const float* __restrict__ g,
                                                             const float* __restrict__ b,
                                                             const float* __restrict__ gsum,
                                                             const float* __restrict__ gsumsq,
                                                             float* __restrict__ scale,
                                                             float* __restrict__ shift,
                                                             int cout) {
    const int t = threadIdx.x;
    if (t < cout) {
        const float inv = 1.0f / (float)NK;
        const float mean = gsum[t] * inv;
        const float var = gsumsq[t] * inv - mean * mean;
        const float sc = g[t] * rsqrtf(var + BN_EPS);
        scale[t] = sc;
        shift[t] = b[t] - mean * sc;
    }
}

// ============================================================
// BN + LeakyReLU -> bf16 next-layer input
// ============================================================
__global__ __launch_bounds__(256) void bn_act_kernel(const float* __restrict__ Y,
                                                     const float* __restrict__ scale,
                                                     const float* __restrict__ shift,
                                                     __hip_bfloat16* __restrict__ X,
                                                     int cout) {
    const size_t i = (size_t)blockIdx.x * 256 + threadIdx.x;
    const int c = (int)(i & (size_t)(cout - 1));
    float y = Y[i] * scale[c] + shift[c];
    y = (y >= 0.0f) ? y : SLOPE * y;
    X[i] = __float2bfloat16(y);
}

// ============================================================
// Final layer: BN + LeakyReLU + max over K -> (N, 256) f32 out
// ============================================================
__global__ __launch_bounds__(256) void bn_act_max_kernel(const float* __restrict__ Y,
                                                         const float* __restrict__ scale,
                                                         const float* __restrict__ shift,
                                                         float* __restrict__ out) {
    const int n = blockIdx.x;
    const int c = threadIdx.x;
    const float sc = scale[c], sh = shift[c];
    float m = -3.402823466e38f;
#pragma unroll
    for (int k = 0; k < KNN_K; ++k) {
        float y = Y[(size_t)(n * KNN_K + k) * 256 + c] * sc + sh;
        y = (y >= 0.0f) ? y : SLOPE * y;
        m = fmaxf(m, y);
    }
    out[(size_t)n * 256 + c] = m;
}

__global__ __launch_bounds__(256) void copy_pos1_kernel(const float* __restrict__ pos1,
                                                        float* __restrict__ out) {
    const int i = blockIdx.x * 256 + threadIdx.x;
    out[i] = pos1[i];
}

// ============================================================
extern "C" void kernel_launch(void* const* d_in, const int* in_sizes, int n_in,
                              void* d_out, int out_size, void* d_ws, size_t ws_size,
                              hipStream_t stream) {
    const float* pos1  = (const float*)d_in[0];
    const float* pos2  = (const float*)d_in[1];
    const float* feat1 = (const float*)d_in[2];
    const float* feat2 = (const float*)d_in[3];
    const float* W0 = (const float*)d_in[6];
    const float* g0 = (const float*)d_in[7];
    const float* b0 = (const float*)d_in[8];
    const float* W1 = (const float*)d_in[9];
    const float* g1 = (const float*)d_in[10];
    const float* b1 = (const float*)d_in[11];
    const float* W2 = (const float*)d_in[12];
    const float* g2 = (const float*)d_in[13];
    const float* b2 = (const float*)d_in[14];

    char* ws = (char*)d_ws;
    int*            idxb   = (int*)(ws + OFF_IDX);
    float*          gsum   = (float*)(ws + OFF_STATS);
    float*          gsumsq = gsum + 256;
    float*          scale  = gsum + 512;
    float*          shift  = gsum + 768;
    __hip_bfloat16* Wb     = (__hip_bfloat16*)(ws + OFF_W);
    __hip_bfloat16* Xb     = (__hip_bfloat16*)(ws + OFF_X);
    float*          Yb     = (float*)(ws + OFF_Y);
    float*          out    = (float*)d_out;

    // 1) KNN + gather/concat into bf16 GEMM input
    knn_kernel<<<NPTS / 256, 256, 0, stream>>>(pos1, pos2, idxb);
    build_x0_kernel<<<(NK * KPAD0) / 256, 256, 0, stream>>>(pos1, pos2, feat1, feat2, idxb, Xb);

    // 2) Layer 0: (NK,288) x (128,288)^T
    convert_w_kernel<<<(128 * KPAD0) / 256, 256, 0, stream>>>(W0, Wb, CIN0, KPAD0);
    zero_stats_kernel<<<1, 256, 0, stream>>>(gsum, gsumsq);
    gemm_bf16_kernel<9, 128><<<dim3(NK / 64, 1), 256, 0, stream>>>(
        (const u32x4*)Xb, (const u32x4*)Wb, Yb);
    stats_kernel<<<256, 256, 0, stream>>>(Yb, gsum, gsumsq, 128);
    finalize_stats_kernel<<<1, 256, 0, stream>>>(g0, b0, gsum, gsumsq, scale, shift, 128);
    bn_act_kernel<<<(NK * 128) / 256, 256, 0, stream>>>(Yb, scale, shift, Xb, 128);

    // 3) Layer 1: (NK,128) x (128,128)^T
    convert_w_kernel<<<(128 * 128) / 256, 256, 0, stream>>>(W1, Wb, 128, 128);
    zero_stats_kernel<<<1, 256, 0, stream>>>(gsum, gsumsq);
    gemm_bf16_kernel<4, 128><<<dim3(NK / 64, 1), 256, 0, stream>>>(
        (const u32x4*)Xb, (const u32x4*)Wb, Yb);
    stats_kernel<<<256, 256, 0, stream>>>(Yb, gsum, gsumsq, 128);
    finalize_stats_kernel<<<1, 256, 0, stream>>>(g1, b1, gsum, gsumsq, scale, shift, 128);
    bn_act_kernel<<<(NK * 128) / 256, 256, 0, stream>>>(Yb, scale, shift, Xb, 128);

    // 4) Layer 2: (NK,128) x (256,128)^T, then BN+LReLU+max over K
    convert_w_kernel<<<(256 * 128) / 256, 256, 0, stream>>>(W2, Wb, 128, 128);
    zero_stats_kernel<<<1, 256, 0, stream>>>(gsum, gsumsq);
    gemm_bf16_kernel<4, 256><<<dim3(NK / 64, 2), 256, 0, stream>>>(
        (const u32x4*)Xb, (const u32x4*)Wb, Yb);
    stats_kernel<<<256, 256, 0, stream>>>(Yb, gsum, gsumsq, 256);
    finalize_stats_kernel<<<1, 256, 0, stream>>>(g2, b2, gsum, gsumsq, scale, shift, 256);
    bn_act_max_kernel<<<NPTS, 256, 0, stream>>>(Yb, scale, shift, out + NPTS * 3);

    // 5) pos1 passthrough (first tuple element)
    copy_pos1_kernel<<<(NPTS * 3) / 256, 256, 0, stream>>>(pos1, out);
}
